// ClusterAttention_Old_14620068675736
// MI455X (gfx1250) — matrix operand order repaired and checked by reading, not compiled, not run on hardware
//
#include <hip/hip_runtime.h>

typedef __attribute__((ext_vector_type(16))) __bf16 v16bf;
typedef __attribute__((ext_vector_type(8)))  __bf16 v8bf;
typedef __attribute__((ext_vector_type(8)))  float  v8f;

#define BB 4
#define NN 1024
#define CC 512
#define HH 16
#define CD 32
#define TT 4096

static __device__ __forceinline__ v8f zero8() {
  v8f z;
#pragma unroll
  for (int i = 0; i < 8; ++i) z[i] = 0.0f;
  return z;
}

// Build a 16-element A-operand vector from two contiguous 8-element (16B) chunks.
// A-layout as callers use it (16-bit A 16x32): lane<16 holds row=lane, K in {0..15};
// lane>=16 holds row=lane-16, K in {16..31}. Elements 0..7 = first run,
// 8..15 = second run (2 bf16 per VGPR).
static __device__ __forceinline__ v16bf ld_split8(const __bf16* p0, const __bf16* p1) {
  v8bf x = *(const v8bf*)p0;
  v8bf y = *(const v8bf*)p1;
  v16bf r;
#pragma unroll
  for (int i = 0; i < 8; ++i) { r[i] = x[i]; r[8 + i] = y[i]; }
  return r;
}

static __device__ __forceinline__ v8f wmma_bf16(v16bf a, v16bf b, v8f c) {
  return __builtin_amdgcn_wmma_f32_16x16x32_bf16(false, a, false, b, (short)0, c, false, false);
}

// ---------------- elementwise f32 -> bf16 ----------------
__global__ __launch_bounds__(256) void cvt_bf16_kernel(const float* __restrict__ src,
                                                       __bf16* __restrict__ dst, int count) {
  int i = blockIdx.x * 256 + threadIdx.x;
  if (i < count) dst[i] = (__bf16)src[i];
}

// ---------------- pe_out[t][j] = pre_table[t,:] . pe_w[j,:] + pe_b[j] ----------------
__global__ __launch_bounds__(256) void pe_kernel(const float* __restrict__ pre_table,
                                                 const float* __restrict__ pe_w,
                                                 const float* __restrict__ pe_b,
                                                 float* __restrict__ pe_out) {
  int i = blockIdx.x * 256 + threadIdx.x;  // over T*H
  if (i >= TT * HH) return;
  int t = i >> 4, j = i & 15;
  float s = pe_b[j];
#pragma unroll
  for (int k = 0; k < 5; ++k) s += pre_table[t * 5 + k] * pe_w[j * 5 + k];
  pe_out[i] = s;
}

// ---------------- 16x64 GEMM tile per wave, A: MxK row-major, B: NxK row-major (out col = B row)
static __device__ __forceinline__ void gemm16x64_k512(const __bf16* __restrict__ A,
                                                      const __bf16* __restrict__ B,
                                                      int mBase, int nBase, v8f acc[4]) {
  int lane = threadIdx.x & 31;
  int lo = lane & 15, hi = lane >> 4;
  const __bf16* arow = A + (size_t)(mBase + lo) * 512;
#pragma unroll 1
  for (int kk = 0; kk < 512; kk += 32) {
    v16bf a = ld_split8(arow + kk + hi * 16, arow + kk + hi * 16 + 8);
#pragma unroll
    for (int j = 0; j < 4; ++j) {
      // B-layout: lane<16 holds col=nBase+j*16+lane, K=kk..kk+15 contiguous; lane>=16 K=kk+16..kk+31
      const __bf16* brow = B + (size_t)(nBase + j * 16 + lo) * 512 + kk + hi * 16;
      v16bf b = *(const v16bf*)brow;
      acc[j] = wmma_bf16(a, b, acc[j]);
    }
  }
}

// ---------------- QKV projection: feat_bf (4096x512) x Wqkv^T (1536 rows) ----------------
__global__ __launch_bounds__(128) void qkv_kernel(const __bf16* __restrict__ featb,
                                                  const __bf16* __restrict__ wqkv,
                                                  const float* __restrict__ q_b,
                                                  const float* __restrict__ kv_b,
                                                  __bf16* __restrict__ qo,   // (b,h,n,32), pre-scaled
                                                  __bf16* __restrict__ ko,   // (b,h,n,32)
                                                  __bf16* __restrict__ vt) { // (b,h,32,n)  transposed
  const float scale = 0.17677669529663687f;  // 32^-0.5
  const int NW = 1536 / 64;
  int waveId = blockIdx.x * 4 + (threadIdx.x >> 5);
  int mTile = waveId / NW, nW = waveId % NW;
  int mBase = mTile * 16, nBase = nW * 64;
  v8f acc[4];
#pragma unroll
  for (int j = 0; j < 4; ++j) acc[j] = zero8();
  gemm16x64_k512(featb, wqkv, mBase, nBase, acc);
  int lane = threadIdx.x & 31, lo = lane & 15, hi = lane >> 4;
#pragma unroll
  for (int j = 0; j < 4; ++j) {
    int col = nBase + j * 16 + lo;
#pragma unroll
    for (int r = 0; r < 8; ++r) {
      int row = mBase + r + hi * 8;           // C-layout: VGPR r -> M=r (lanes<16) / r+8
      int b = row >> 10, nrow = row & 1023;
      float v = acc[j][r];
      if (col < 512) {
        int h = col >> 5, d = col & 31;
        qo[(((size_t)b * 16 + h) * 1024 + nrow) * 32 + d] = (__bf16)((v + q_b[col]) * scale);
      } else {
        int g = col - 512;
        int h = g >> 6, rem = g & 63, d = rem & 31;
        float x = v + kv_b[g];
        if (rem < 32) ko[(((size_t)b * 16 + h) * 1024 + nrow) * 32 + d] = (__bf16)x;
        else          vt[(((size_t)b * 16 + h) * 32 + d) * 1024 + nrow] = (__bf16)x;
      }
    }
  }
}

// ---------------- flash attention: block = (b, h, 64 q-rows); 4 waves x 16 q-rows ----------------
__global__ __launch_bounds__(128) void attn_kernel(const __bf16* __restrict__ qb,
                                                   const __bf16* __restrict__ kb,
                                                   const __bf16* __restrict__ vt,
                                                   const int* __restrict__ pe_idx,
                                                   const float* __restrict__ pe_out,
                                                   const float* __restrict__ blank_k,
                                                   const float* __restrict__ blank_v,
                                                   __bf16* __restrict__ ao) {
  __shared__ float peh[TT];            // this head's positional table column (16 KB)
  __shared__ __bf16 pst[4][16 * 32];   // per-wave P staging (C-layout -> A-layout via LDS)

  int blk = blockIdx.x;                // 4*16*16 = 1024
  int qB = blk & 15;
  int h = (blk >> 4) & 15;
  int bi = blk >> 8;
  for (int t = threadIdx.x; t < TT; t += 128) peh[t] = pe_out[t * 16 + h];
  __syncthreads();

  int w = threadIdx.x >> 5, lane = threadIdx.x & 31, lo = lane & 15, hi = lane >> 4;
  int qBase = qB * 64 + w * 16;
  const __bf16* qhead = qb + (((size_t)bi * 16 + h) * 1024) * 32;
  const __bf16* khead = kb + (((size_t)bi * 16 + h) * 1024) * 32;
  const __bf16* vhead = vt + (((size_t)bi * 16 + h) * 32) * 1024;

  // Q tile in A-layout, reused for all 32 K-tiles
  const __bf16* qr = qhead + (size_t)(qBase + lo) * 32;
  v16bf aq = ld_split8(qr + hi * 16, qr + hi * 16 + 8);

  // blank logit for row lo (q already scaled)
  float bl = 0.0f;
#pragma unroll
  for (int d = 0; d < 32; ++d) bl += (float)qr[d] * blank_k[h * 32 + d];

  float mst[8], lst[8], bls[8];
  v8f acc0 = zero8(), acc1 = zero8();
#pragma unroll
  for (int r = 0; r < 8; ++r) {
    int src = hi ? (24 + r) : r;       // lane that computed blank for row r + hi*8
    float b0 = __shfl(bl, src, 32);
    bls[r] = b0; mst[r] = b0; lst[r] = 1.0f;   // blank column seeds running softmax
  }
  const int* perow_base = pe_idx + (size_t)bi * 1024 * 1024;

#pragma unroll 1
  for (int k0 = 0; k0 < 1024; k0 += 32) {
    v16bf bk0 = *(const v16bf*)(khead + (size_t)(k0 + lo) * 32 + hi * 16);
    v16bf bk1 = *(const v16bf*)(khead + (size_t)(k0 + 16 + lo) * 32 + hi * 16);
    v8f s0 = wmma_bf16(aq, bk0, zero8());
    v8f s1 = wmma_bf16(aq, bk1, zero8());

    // add positional bias: gather pe index (global, L2-resident), look up in LDS
#pragma unroll
    for (int r = 0; r < 8; ++r) {
      int row = qBase + r + hi * 8;
      const int* pr = perow_base + (size_t)row * 1024 + k0;
      s0[r] += peh[pr[lo]];
      s1[r] += peh[pr[16 + lo]];
    }

    // online softmax (reductions stay within 16-lane halves = one C-layout row group)
#pragma unroll
    for (int r = 0; r < 8; ++r) {
      float mx = fmaxf(s0[r], s1[r]);
#pragma unroll
      for (int off = 1; off < 16; off <<= 1) mx = fmaxf(mx, __shfl_xor(mx, off, 32));
      float mnew = fmaxf(mst[r], mx);
      float corr = __expf(mst[r] - mnew);
      float p0 = __expf(s0[r] - mnew);
      float p1 = __expf(s1[r] - mnew);
      float ps = p0 + p1;
#pragma unroll
      for (int off = 1; off < 16; off <<= 1) ps += __shfl_xor(ps, off, 32);
      lst[r] = lst[r] * corr + ps;
      mst[r] = mnew;
      acc0[r] *= corr;
      acc1[r] *= corr;
      pst[w][(r + hi * 8) * 32 + lo] = (__bf16)p0;
      pst[w][(r + hi * 8) * 32 + 16 + lo] = (__bf16)p1;
    }

    // P(16x32) x V(32x32): reload P in A-layout (wave-private LDS, same-wave DS ops are in-order)
    v16bf ap = ld_split8(&pst[w][lo * 32 + hi * 16], &pst[w][lo * 32 + hi * 16 + 8]);
    v16bf v0 = *(const v16bf*)(vhead + (size_t)lo * 1024 + k0 + hi * 16);
    v16bf v1 = *(const v16bf*)(vhead + (size_t)(16 + lo) * 1024 + k0 + hi * 16);
    acc0 = wmma_bf16(ap, v0, acc0);
    acc1 = wmma_bf16(ap, v1, acc1);
  }

  // finalize: normalize, add blank-value contribution, store bf16 (b,n,c)
#pragma unroll
  for (int r = 0; r < 8; ++r) {
    float inv = 1.0f / lst[r];
    float bw = __expf(bls[r] - mst[r]) * inv;
    int row = qBase + r + hi * 8;
    float o0 = acc0[r] * inv + bw * blank_v[h * 32 + lo];
    float o1 = acc1[r] * inv + bw * blank_v[h * 32 + 16 + lo];
    __bf16* orow = ao + ((size_t)bi * 1024 + row) * 512 + h * 32;
    orow[lo] = (__bf16)o0;
    orow[16 + lo] = (__bf16)o1;
  }
}

// ---------------- output projection: attn_out_bf (4096x512) x proj_w^T + proj_b -> f32 ----------------
__global__ __launch_bounds__(128) void proj_kernel(const __bf16* __restrict__ ab,
                                                   const __bf16* __restrict__ wp,
                                                   const float* __restrict__ proj_b,
                                                   float* __restrict__ out) {
  const int NW = 512 / 64;
  int waveId = blockIdx.x * 4 + (threadIdx.x >> 5);
  int mTile = waveId / NW, nW = waveId % NW;
  int mBase = mTile * 16, nBase = nW * 64;
  v8f acc[4];
#pragma unroll
  for (int j = 0; j < 4; ++j) acc[j] = zero8();
  gemm16x64_k512(ab, wp, mBase, nBase, acc);
  int lane = threadIdx.x & 31, lo = lane & 15, hi = lane >> 4;
#pragma unroll
  for (int j = 0; j < 4; ++j) {
    int col = nBase + j * 16 + lo;
#pragma unroll
    for (int r = 0; r < 8; ++r) {
      int row = mBase + r + hi * 8;
      out[(size_t)row * 512 + col] = acc[j][r] + proj_b[col];
    }
  }
}

extern "C" void kernel_launch(void* const* d_in, const int* in_sizes, int n_in,
                              void* d_out, int out_size, void* d_ws, size_t ws_size,
                              hipStream_t stream) {
  const float* feat      = (const float*)d_in[0];
  const int*   pe_idx    = (const int*)d_in[3];
  const float* pre_table = (const float*)d_in[5];
  const float* q_w       = (const float*)d_in[6];
  const float* q_b       = (const float*)d_in[7];
  const float* kv_w      = (const float*)d_in[8];
  const float* kv_b      = (const float*)d_in[9];
  const float* pe_w      = (const float*)d_in[10];
  const float* pe_b      = (const float*)d_in[11];
  const float* blank_k   = (const float*)d_in[12];
  const float* blank_v   = (const float*)d_in[13];
  const float* proj_w    = (const float*)d_in[14];
  const float* proj_b    = (const float*)d_in[15];
  float* out = (float*)d_out;

  char* ws = (char*)d_ws;
  size_t off = 0;
  auto salloc = [&](size_t bytes) {
    char* p = ws + off;
    off += (bytes + 255) & ~(size_t)255;
    return (void*)p;
  };
  __bf16* featb = (__bf16*)salloc((size_t)4096 * 512 * 2);
  __bf16* wqkvb = (__bf16*)salloc((size_t)1536 * 512 * 2);  // q_w rows then kv_w rows
  __bf16* wpb   = (__bf16*)salloc((size_t)512 * 512 * 2);
  __bf16* qbf   = (__bf16*)salloc((size_t)BB * HH * NN * CD * 2);
  __bf16* kbf   = (__bf16*)salloc((size_t)BB * HH * NN * CD * 2);
  __bf16* vtb   = (__bf16*)salloc((size_t)BB * HH * CD * NN * 2);
  float*  peo   = (float*)salloc((size_t)TT * HH * 4);
  __bf16* aob   = (__bf16*)salloc((size_t)4096 * 512 * 2);

  // conversions to bf16
  cvt_bf16_kernel<<<(4096 * 512 + 255) / 256, 256, 0, stream>>>(feat, featb, 4096 * 512);
  cvt_bf16_kernel<<<(512 * 512 + 255) / 256, 256, 0, stream>>>(q_w, wqkvb, 512 * 512);
  cvt_bf16_kernel<<<(1024 * 512 + 255) / 256, 256, 0, stream>>>(kv_w, wqkvb + (size_t)512 * 512, 1024 * 512);
  cvt_bf16_kernel<<<(512 * 512 + 255) / 256, 256, 0, stream>>>(proj_w, wpb, 512 * 512);

  // positional table
  pe_kernel<<<(TT * HH + 255) / 256, 256, 0, stream>>>(pre_table, pe_w, pe_b, peo);

  // QKV projection (M=4096, N=1536, K=512): 256*24 waves / 4 per block
  qkv_kernel<<<1536, 128, 0, stream>>>(featb, wqkvb, q_b, kv_b, qbf, kbf, vtb);

  // flash attention: one block per (b, h, 64 q rows)
  attn_kernel<<<BB * HH * (NN / 64), 128, 0, stream>>>(qbf, kbf, vtb, pe_idx, peo,
                                                       blank_k, blank_v, aob);

  // output projection (M=4096, N=512, K=512): 256*8 waves / 4 per block
  proj_kernel<<<512, 128, 0, stream>>>(aob, wpb, proj_b, out);
}
